// CBDAE_8693013807684
// MI455X (gfx1250) — compile-verified, matrix operand order checked by reading, not run on hardware
//
#include <hip/hip_runtime.h>
#include <stdint.h>

// Problem constants (match the reference)
#define B_   128
#define T_   512
#define N_   128
#define H_   512
#define G1_  256
#define G_   128
#define THREE_H (3 * H_)

typedef __bf16 bf16_t;
typedef __attribute__((ext_vector_type(8)))  __bf16 v8bf;
typedef __attribute__((ext_vector_type(16))) __bf16 v16bf;
typedef __attribute__((ext_vector_type(8)))  float  v8f;

// LDS byte layout inside gru_step_kernel (dynamic shared):
//   Whh slab: 48 rows, row stride (H_+8) bf16 = 1040 B  -> 49920 B
//   Wih slab (only when WIH_LDS): 48 rows, stride (KX+8) bf16
#define WHH_ROW_E   (H_ + 8)              // 520 elements
#define WHH_SLAB_B  (48 * WHH_ROW_E * 2)  // 49920 bytes
#define WIH_BASE_E  (WHH_SLAB_B / 2)      // element offset of Wih slab

// ---------------------------------------------------------------------------
// Fragment loader for V_WMMA_F32_16X16X32_BF16 (A or B^T, row-major source).
// Lanes 0-15 (row = l15) hold K = k0..k0+7 and k0+16..k0+23; lanes 16-31 hold
// K = k0+8..15 and k0+24..31 -> kb = k0 + (lane>>4)*8, two 16B loads.
// ---------------------------------------------------------------------------
__device__ __forceinline__ v16bf load_frag(const bf16_t* __restrict__ base,
                                           int row, int ld, int kb) {
  const bf16_t* p = base + row * ld + kb;
  v8bf lo = *(const v8bf*)(p);
  v8bf hi = *(const v8bf*)(p + 16);
  v16bf r;
#pragma unroll
  for (int i = 0; i < 8; ++i) { r[i] = lo[i]; r[i + 8] = hi[i]; }
  return r;
}

// Same fragment shape but from an LDS-resident slab pointer (lowers to ds_load).
__device__ __forceinline__ v16bf load_frag_s(const bf16_t* p) {
  v8bf lo = *(const v8bf*)(p);
  v8bf hi = *(const v8bf*)(p + 16);
  v16bf r;
#pragma unroll
  for (int i = 0; i < 8; ++i) { r[i] = lo[i]; r[i + 8] = hi[i]; }
  return r;
}

__device__ __forceinline__ v8f wmma_bf16(v16bf a, v16bf b, v8f c) {
  return __builtin_amdgcn_wmma_f32_16x16x32_bf16(false, a, false, b,
                                                 (short)0, c, false, false);
}

// Async global -> LDS 16-byte copy (per active lane), tracked by ASYNCcnt.
__device__ __forceinline__ void async_g2l_b128(uint32_t lds_byte_off,
                                               const void* gptr) {
  asm volatile("global_load_async_to_lds_b128 %0, %1, off"
               :: "v"(lds_byte_off),
                  "v"((unsigned long long)(uintptr_t)gptr)
               : "memory");
}
__device__ __forceinline__ void wait_async0() {
  asm volatile("s_wait_asynccnt 0x0" ::: "memory");
}

__device__ __forceinline__ float sigmoidf_(float x) {
  return 1.0f / (1.0f + __expf(-x));
}

// ---------------------------------------------------------------------------
// Fully fused GRU cell step:  h_out = GRU(x_t, h_in)
// One wave owns a 16x16 tile of (B=128) x (H=512); 6 f32 WMMA accumulators
// (r/z/n for gi and gh). All 8 waves of a block share the same jt, so the
// block's weight slab is staged once into LDS via async loads; B fragments
// are then served from LDS (ds_load_b128), A fragments stream from global.
// KX / WIH_LDS are compile-time so each instantiation has straight-line,
// fully pipelinable K-loops (no in-loop branches, no accumulator copies).
// Grid = 32 blocks x 256 threads.
// ---------------------------------------------------------------------------
template <int KX, bool WIH_LDS>
__global__ __launch_bounds__(256)
void gru_step_kernel(const bf16_t* __restrict__ X, int ldx,
                     const bf16_t* __restrict__ Wih,   // (3H, KX) bf16
                     const bf16_t* __restrict__ Whh,   // (3H, H)  bf16
                     const float*  __restrict__ bih,   // (3H) f32
                     const float*  __restrict__ bhh,   // (3H) f32
                     const bf16_t* __restrict__ Hin_bf,// (B, H) bf16
                     const float*  __restrict__ Hin_f, // (B, H) f32
                     bf16_t* __restrict__ Hout_bf,
                     float*  __restrict__ Hout_f) {
  extern __shared__ bf16_t smem[];

  const int tid  = threadIdx.x;
  const int lane = tid & 31;
  const int wid  = (blockIdx.x * blockDim.x + tid) >> 5; // 0..255
  const int mt   = wid & 7;    // 8 tiles over B=128
  const int jt   = wid >> 3;   // 32 tiles over H=512 (uniform within block)
  const int half = lane >> 4;
  const int l15  = lane & 15;

  const int arow  = mt * 16 + l15;
  const int kboff = half * 8;

  const uint32_t lds_base = (uint32_t)(uintptr_t)smem; // LDS byte offset

  // ---- stage Whh slab: 48 rows (3 gates x 16 cols) x H_, padded rows ----
  {
    constexpr unsigned CPR = H_ / 8;      // 64 x 16B chunks per row
#pragma unroll 1
    for (unsigned c = tid; c < 48u * CPR; c += 256u) {
      unsigned r = c >> 6, w = c & (CPR - 1u);
      int wrow = (int)(r >> 4) * H_ + jt * 16 + (int)(r & 15u);
      async_g2l_b128(lds_base + r * (WHH_ROW_E * 2) + w * 16u,
                     Whh + (size_t)wrow * H_ + w * 8u);
    }
  }
  // ---- stage Wih slab only for small KX (keeps dynamic LDS < 64KB) ----
  if constexpr (WIH_LDS) {
    constexpr unsigned CPR = (unsigned)KX / 8;
#pragma unroll 1
    for (unsigned c = tid; c < 48u * CPR; c += 256u) {
      unsigned r = c / CPR, w = c & (CPR - 1u);
      int wrow = (int)(r >> 4) * H_ + jt * 16 + (int)(r & 15u);
      async_g2l_b128(lds_base + (uint32_t)WHH_SLAB_B +
                         r * (uint32_t)((KX + 8) * 2) + w * 16u,
                     Wih + (size_t)wrow * KX + w * 8u);
    }
  }
  wait_async0();
  __syncthreads();

  const bf16_t* whh_s = smem;               // 48 x WHH_ROW_E
  const bf16_t* wih_s = smem + WIH_BASE_E;  // 48 x (KX+8)

  const v8f zero = {0.f, 0.f, 0.f, 0.f, 0.f, 0.f, 0.f, 0.f};
  v8f acc[6];
#pragma unroll
  for (int g = 0; g < 6; ++g) acc[g] = zero;

  // gi accumulation: x @ Wih^T — single straight-line loop per instantiation
  if constexpr (WIH_LDS) {
    constexpr int ROWE = KX + 8;
#pragma unroll
    for (int k0 = 0; k0 < KX; k0 += 32) {
      v16bf a = load_frag(X, arow, ldx, k0 + kboff);
#pragma unroll
      for (int g = 0; g < 3; ++g) {
        v16bf b = load_frag_s(wih_s + (g * 16 + l15) * ROWE + k0 + kboff);
        acc[g] = wmma_bf16(a, b, acc[g]);
      }
    }
  } else {
    for (int k0 = 0; k0 < KX; k0 += 32) {
      v16bf a = load_frag(X, arow, ldx, k0 + kboff);
#pragma unroll
      for (int g = 0; g < 3; ++g) {
        int wrow = g * H_ + jt * 16 + l15;
        v16bf b = load_frag(Wih, wrow, KX, k0 + kboff);
        acc[g] = wmma_bf16(a, b, acc[g]);
      }
    }
  }
  // gh accumulation: h @ Whh^T (B fragments from LDS)
  for (int k0 = 0; k0 < H_; k0 += 32) {
    v16bf a = load_frag(Hin_bf, arow, H_, k0 + kboff);
#pragma unroll
    for (int g = 0; g < 3; ++g) {
      v16bf b = load_frag_s(whh_s + (g * 16 + l15) * WHH_ROW_E + k0 + kboff);
      acc[3 + g] = wmma_bf16(a, b, acc[3 + g]);
    }
  }

  // Gate math in-register. C/D layout: vgpr r -> (m = mt*16 + half*8 + r,
  // n = jt*16 + l15).
  const int col = jt * 16 + l15;
  const float bir  = bih[col],          bhr = bhh[col];
  const float biz  = bih[H_ + col],     bhz = bhh[H_ + col];
  const float bin_ = bih[2 * H_ + col], bhn = bhh[2 * H_ + col];
#pragma unroll
  for (int r = 0; r < 8; ++r) {
    int   m  = mt * 16 + half * 8 + r;
    float hp = Hin_f[m * H_ + col];
    float rr = sigmoidf_(acc[0][r] + bir + acc[3][r] + bhr);
    float zz = sigmoidf_(acc[1][r] + biz + acc[4][r] + bhz);
    float nn = tanhf(acc[2][r] + bin_ + rr * (acc[5][r] + bhn));
    float hn = (1.0f - zz) * nn + zz * hp;
    Hout_f[m * H_ + col]  = hn;
    Hout_bf[m * H_ + col] = (bf16_t)hn;
  }
}

// ---------------------------------------------------------------------------
// Generic small GEMM: C = act(A_bf16 @ W_bf16^T + bias) — bottleneck MLP and
// decoder output projection. One wave per 16x16 tile.
// ---------------------------------------------------------------------------
__global__ __launch_bounds__(256)
void gemm_bf16_kernel(const bf16_t* __restrict__ A, int lda,
                      const bf16_t* __restrict__ W,     // (Nout, K)
                      const float*  __restrict__ bias,  // nullable
                      int K, int tilesM,
                      float*  __restrict__ Cf,  int ldc,  // nullable
                      bf16_t* __restrict__ Cbf, int ldcb, // nullable
                      int relu) {
  const int lane = threadIdx.x & 31;
  const int wid  = (blockIdx.x * blockDim.x + threadIdx.x) >> 5;
  const int mt   = wid % tilesM;
  const int nt   = wid / tilesM;
  const int half = lane >> 4;
  const int l15  = lane & 15;
  const int arow  = mt * 16 + l15;
  const int wrow  = nt * 16 + l15;
  const int kboff = half * 8;

  v8f acc = {0.f, 0.f, 0.f, 0.f, 0.f, 0.f, 0.f, 0.f};
  for (int k0 = 0; k0 < K; k0 += 32) {
    v16bf a = load_frag(A, arow, lda, k0 + kboff);
    v16bf b = load_frag(W, wrow, K, k0 + kboff);
    acc = wmma_bf16(a, b, acc);
  }

  const int col = nt * 16 + l15;
  const float bv = bias ? bias[col] : 0.0f;
#pragma unroll
  for (int r = 0; r < 8; ++r) {
    int   m = mt * 16 + half * 8 + r;
    float v = acc[r] + bv;
    if (relu) v = fmaxf(v, 0.0f);
    if (Cf)  Cf[m * ldc + col]   = v;
    if (Cbf) Cbf[m * ldcb + col] = (bf16_t)v;
  }
}

// f32 -> bf16 conversion (weights / input staging)
__global__ __launch_bounds__(256)
void cvt_f32_bf16(const float* __restrict__ in, bf16_t* __restrict__ out, int n) {
  int i = blockIdx.x * blockDim.x + threadIdx.x;
  if (i < n) out[i] = (bf16_t)in[i];
}

// ---------------------------------------------------------------------------
extern "C" void kernel_launch(void* const* d_in, const int* in_sizes, int n_in,
                              void* d_out, int out_size, void* d_ws, size_t ws_size,
                              hipStream_t stream) {
  (void)in_sizes; (void)n_in; (void)out_size; (void)ws_size;

  const float* x      = (const float*)d_in[0];
  const float* eWih0  = (const float*)d_in[1];
  const float* eWhh0  = (const float*)d_in[2];
  const float* ebih0  = (const float*)d_in[3];
  const float* ebhh0  = (const float*)d_in[4];
  const float* eWih1  = (const float*)d_in[5];
  const float* eWhh1  = (const float*)d_in[6];
  const float* ebih1  = (const float*)d_in[7];
  const float* ebhh1  = (const float*)d_in[8];
  const float* dWih0  = (const float*)d_in[9];
  const float* dWhh0  = (const float*)d_in[10];
  const float* dbih0  = (const float*)d_in[11];
  const float* dbhh0  = (const float*)d_in[12];
  const float* dWih1  = (const float*)d_in[13];
  const float* dWhh1  = (const float*)d_in[14];
  const float* dbih1  = (const float*)d_in[15];
  const float* dbhh1  = (const float*)d_in[16];
  const float* Wg1    = (const float*)d_in[17];
  const float* Wg2    = (const float*)d_in[18];
  const float* Woutw  = (const float*)d_in[19];
  const float* Woutb  = (const float*)d_in[20];

  float* outf = (float*)d_out;  // [zT (B*G)] ++ [y_hat (B*T*N)]

  // ---- workspace carve-out (256B aligned) ----
  char* ws = (char*)d_ws;
  size_t off = 0;
  auto alloc = [&](size_t bytes) -> void* {
    off = (off + 255) & ~(size_t)255;
    void* p = ws + off;
    off += bytes;
    return p;
  };

  bf16_t* xb = (bf16_t*)alloc((size_t)B_ * T_ * N_ * sizeof(bf16_t));

  const float* wsrc[11] = {eWih0, eWhh0, eWih1, eWhh1,
                           dWih0, dWhh0, dWih1, dWhh1,
                           Wg1,   Wg2,   Woutw};
  const int wcnt[11] = {THREE_H * N_, THREE_H * H_, THREE_H * H_, THREE_H * H_,
                        THREE_H * N_, THREE_H * H_, THREE_H * H_, THREE_H * H_,
                        G1_ * H_, G_ * G1_, N_ * H_};
  bf16_t* wb[11];
  for (int i = 0; i < 11; ++i)
    wb[i] = (bf16_t*)alloc((size_t)wcnt[i] * sizeof(bf16_t));

  // double-buffered states: s0 = h1/d0 , s1 = h2/d1
  float*  s0f[2];  bf16_t* s0b[2];
  float*  s1f[2];  bf16_t* s1b[2];
  for (int i = 0; i < 2; ++i) {
    s0f[i] = (float*) alloc((size_t)B_ * H_ * sizeof(float));
    s0b[i] = (bf16_t*)alloc((size_t)B_ * H_ * sizeof(bf16_t));
    s1f[i] = (float*) alloc((size_t)B_ * H_ * sizeof(float));
    s1b[i] = (bf16_t*)alloc((size_t)B_ * H_ * sizeof(bf16_t));
  }
  bf16_t* g1b = (bf16_t*)alloc((size_t)B_ * G1_ * sizeof(bf16_t));
  bf16_t* yb  = (bf16_t*)alloc((size_t)B_ * N_ * sizeof(bf16_t));

  // dynamic LDS sizes for the fused GRU kernel
  const size_t shmem_small = (size_t)WHH_SLAB_B + 48u * (N_ + 8) * 2; // KX=128
  const size_t shmem_big   = (size_t)WHH_SLAB_B;                     // KX=512

  // ---- staging: bf16 conversions + zero-init (all stream-ordered) ----
  {
    int n = B_ * T_ * N_;
    cvt_f32_bf16<<<(n + 255) / 256, 256, 0, stream>>>(x, xb, n);
  }
  for (int i = 0; i < 11; ++i)
    cvt_f32_bf16<<<(wcnt[i] + 255) / 256, 256, 0, stream>>>(wsrc[i], wb[i], wcnt[i]);

  hipMemsetAsync(s0f[0], 0, (size_t)B_ * H_ * sizeof(float),  stream);
  hipMemsetAsync(s0b[0], 0, (size_t)B_ * H_ * sizeof(bf16_t), stream);
  hipMemsetAsync(s1f[0], 0, (size_t)B_ * H_ * sizeof(float),  stream);
  hipMemsetAsync(s1b[0], 0, (size_t)B_ * H_ * sizeof(bf16_t), stream);
  hipMemsetAsync(yb,     0, (size_t)B_ * N_ * sizeof(bf16_t), stream);

  // ---- encoder: both layers advanced inside one time loop ----
  for (int t = 0; t < T_; ++t) {
    int ri = t & 1, wi = ri ^ 1;
    gru_step_kernel<N_, true><<<32, 256, shmem_small, stream>>>(
        xb + (size_t)t * N_, T_ * N_,
        wb[0], wb[1], ebih0, ebhh0,
        s0b[ri], s0f[ri], s0b[wi], s0f[wi]);
    gru_step_kernel<H_, false><<<32, 256, shmem_big, stream>>>(
        s0b[wi], H_,
        wb[2], wb[3], ebih1, ebhh1,
        s1b[ri], s1f[ri], s1b[wi], s1f[wi]);
  }
  // T even: final h1 in s0*[0], final h2 in s1*[0] — decoder reuses in place.

  // ---- bottleneck: zT = relu(h2 @ Wg1^T) @ Wg2^T ----
  gemm_bf16_kernel<<<16, 256, 0, stream>>>(s1b[0], H_, wb[8], nullptr,
                                           H_, 8, nullptr, 0, g1b, G1_, 1);
  gemm_bf16_kernel<<<8, 256, 0, stream>>>(g1b, G1_, wb[9], nullptr,
                                          G1_, 8, outf, G_, nullptr, 0, 0);

  // ---- decoder: autoregressive, y feedback in bf16 ----
  float* yhat = outf + (size_t)B_ * G_;
  for (int t = 0; t < T_; ++t) {
    int ri = t & 1, wi = ri ^ 1;
    gru_step_kernel<N_, true><<<32, 256, shmem_small, stream>>>(
        yb, N_,
        wb[4], wb[5], dbih0, dbhh0,
        s0b[ri], s0f[ri], s0b[wi], s0f[wi]);
    gru_step_kernel<H_, false><<<32, 256, shmem_big, stream>>>(
        s0b[wi], H_,
        wb[6], wb[7], dbih1, dbhh1,
        s1b[ri], s1f[ri], s1b[wi], s1f[wi]);
    gemm_bf16_kernel<<<8, 256, 0, stream>>>(
        s1b[wi], H_, wb[10], Woutb,
        H_, 8, yhat + (size_t)t * N_, T_ * N_, yb, N_, 0);
  }
}
